// TransformerMatchModule_52175262712462
// MI455X (gfx1250) — matrix operand order repaired and checked by reading, not compiled, not run on hardware
//
#include <hip/hip_runtime.h>
#include <hip/hip_bf16.h>

typedef __bf16 bf16;
typedef __attribute__((ext_vector_type(16))) __bf16        v16bf;
typedef __attribute__((ext_vector_type(8)))  float         v8f;
typedef __attribute__((ext_vector_type(4)))  unsigned int  v4u;
typedef __attribute__((ext_vector_type(4)))  float         v4f;

// ---------------- problem dimensions ----------------
constexpr int cB = 8, cK = 256, cDET = 128, cHID = 128, cHEAD = 4, cDKH = 32;
constexpr int cLANG = 256, cCHUNK = 32, cNW = 32;
constexpr int cBC    = cB * cCHUNK;   // 256
constexpr int M_PROP = cB * cK;       // 2048
constexpr int M_LANG = cBC * cNW;     // 8192
constexpr int M_FULL = cBC * cK;      // 65536
constexpr float NEGINF = -1e9f;

// ---------------- WMMA fragment helpers ----------------
union FragBF { v16bf v; v4u q[2]; bf16 h[16]; };

__device__ inline v8f wmma_bf16(v16bf a, v16bf b, v8f c) {
  return __builtin_amdgcn_wmma_f32_16x16x32_bf16(false, a, false, b, (short)0, c,
                                                 false, false);
}

// A fragment (16x32, M x K), A row-major with leading dim ld.
// Lane L: M=L%16, halves 0..15 -> K=kg*16+h (contiguous)
__device__ inline v16bf load_a_frag(const bf16* __restrict__ A, int row0, int k0, int ld) {
  int lane = threadIdx.x & 31;
  int m  = row0 + (lane & 15);
  int kg = lane >> 4;
  const bf16* p = A + (size_t)m * ld + k0;
  FragBF f;
  f.q[0] = *(const v4u*)(p + kg * 16);
  f.q[1] = *(const v4u*)(p + kg * 16 + 8);
  return f.v;
}

// B fragment (32x16, K x N) where memory holds B^T = W (N rows x K cols) row-major.
// Lane = N column; halves 0..15 -> K = kg*16 + h (contiguous in memory).
__device__ inline v16bf load_b_fragT(const bf16* __restrict__ W, int n0, int k0, int ld) {
  int lane = threadIdx.x & 31;
  int n  = n0 + (lane & 15);
  int kg = lane >> 4;
  const bf16* p = W + (size_t)n * ld + k0 + kg * 16;
  FragBF f;
  f.q[0] = *(const v4u*)(p);
  f.q[1] = *(const v4u*)(p + 8);
  return f.v;
}

// B fragment where memory holds B directly (K rows x N cols, row-major): per-lane gather.
__device__ inline v16bf load_b_frag_rm(const bf16* __restrict__ Bm, int k0, int n0, int ld) {
  int lane = threadIdx.x & 31;
  int n  = n0 + (lane & 15);
  int kg = lane >> 4;
  FragBF f;
#pragma unroll
  for (int i = 0; i < 16; ++i)
    f.h[i] = Bm[(size_t)(k0 + kg * 16 + i) * ld + n];
  return f.v;
}

// ---------------- generic WMMA GEMM:  out = epilogue(A * W^T + bias) ----------------
enum { EPI_BF16 = 0, EPI_BN_PRELU = 1, EPI_F32_RES = 2, EPI_F32_RELU = 3 };

__global__ void gemm_wmma(const bf16* __restrict__ A, const bf16* __restrict__ W,
                          const float* __restrict__ bias,
                          int M, int N, int Kd,
                          bf16* __restrict__ outb, float* __restrict__ outf,
                          const bf16* __restrict__ resid,
                          const float* __restrict__ bnG, const float* __restrict__ bnB,
                          const float* __restrict__ bnM, const float* __restrict__ bnV,
                          const float* __restrict__ alpha, int alphaStride, int mode)
{
  const int wave = threadIdx.x >> 5;
  const int row0 = blockIdx.x * 64 + (wave & 1) * 32;
  const int col0 = blockIdx.y * 64 + (wave >> 1) * 32;
  const v8f vz = {0.f, 0.f, 0.f, 0.f, 0.f, 0.f, 0.f, 0.f};
  v8f acc[2][2] = {{vz, vz}, {vz, vz}};

  for (int k0 = 0; k0 < Kd; k0 += 32) {
    v16bf a0 = load_a_frag(A, row0,      k0, Kd);
    v16bf a1 = load_a_frag(A, row0 + 16, k0, Kd);
    v16bf b0 = load_b_fragT(W, col0,      k0, Kd);
    v16bf b1 = load_b_fragT(W, col0 + 16, k0, Kd);
    acc[0][0] = wmma_bf16(a0, b0, acc[0][0]);
    acc[0][1] = wmma_bf16(a0, b1, acc[0][1]);
    acc[1][0] = wmma_bf16(a1, b0, acc[1][0]);
    acc[1][1] = wmma_bf16(a1, b1, acc[1][1]);
  }

  const int lane  = threadIdx.x & 31;
  const int nLoc  = lane & 15;
  const int mBase = (lane >> 4) * 8;
#pragma unroll
  for (int fm = 0; fm < 2; ++fm) {
#pragma unroll
    for (int fn = 0; fn < 2; ++fn) {
      int col = col0 + fn * 16 + nLoc;
      float bia = bias ? bias[col] : 0.f;
      float mu = 0.f, rs = 1.f, gm = 1.f, bt = 0.f, al = 0.f;
      if (mode == EPI_BN_PRELU) {
        mu = bnM[col]; rs = rsqrtf(bnV[col] + 1e-5f);
        gm = bnG[col]; bt = bnB[col];
        al = alpha[alphaStride ? col : 0];
      }
#pragma unroll
      for (int r = 0; r < 8; ++r) {
        int row = row0 + fm * 16 + mBase + r;
        float x = acc[fm][fn][r] + bia;
        size_t off = (size_t)row * N + col;
        if (mode == EPI_BF16) {
          outb[off] = (bf16)x;
        } else if (mode == EPI_BN_PRELU) {
          x = (x - mu) * rs * gm + bt;
          outb[off] = (bf16)(x >= 0.f ? x : al * x);
        } else if (mode == EPI_F32_RES) {
          if (resid) x += (float)resid[off];
          outf[off] = x;
        } else {  // EPI_F32_RELU
          outf[off] = x > 0.f ? x : 0.f;
        }
      }
    }
  }
}

// ---------------- WMMA attention (h=4, dk=dv=32, width 128) ----------------
// grid: (seqQ/16, HEAD, nBatch); block: 32 (1 wave)
__global__ void attn_wmma(const bf16* __restrict__ Q, const bf16* __restrict__ Kb,
                          const bf16* __restrict__ Vb, bf16* __restrict__ O,
                          int seqQ, int seqK,
                          const float* __restrict__ dwAdd, int dwDiv,
                          const int* __restrict__ keyMask, float scale)
{
  __shared__ __align__(16) float sS[16][256];
  __shared__ __align__(16) bf16  sP[16][256];
  const int b  = blockIdx.z;
  const int h  = blockIdx.y;
  const int q0 = blockIdx.x * 16;
  const int lane  = threadIdx.x & 31;
  const int nLoc  = lane & 15;
  const int mBase = (lane >> 4) * 8;
  const v8f vz = {0.f, 0.f, 0.f, 0.f, 0.f, 0.f, 0.f, 0.f};

  const bf16* Qh = Q  + (size_t)b * seqQ * cHID + h * cDKH;
  const bf16* Kh = Kb + (size_t)b * seqK * cHID + h * cDKH;
  const bf16* Vh = Vb + (size_t)b * seqK * cHID + h * cDKH;

  // scores: S = (Q * K^T) * scale  (+dw | mask)
  v16bf aQ = load_a_frag(Qh, q0, 0, cHID);
  for (int kt = 0; kt < seqK / 16; ++kt) {
    v16bf bK = load_b_fragT(Kh, kt * 16, 0, cHID);
    v8f s = wmma_bf16(aQ, bK, vz);
#pragma unroll
    for (int r = 0; r < 8; ++r) {
      int m   = mBase + r;
      int key = kt * 16 + nLoc;
      float sv = s[r] * scale;
      if (dwAdd)
        sv += dwAdd[((size_t)(b / dwDiv) * cK + (q0 + m)) * cK + key];
      if (keyMask && !keyMask[b * seqK + key]) sv = NEGINF;
      sS[m][key] = sv;
    }
  }
  __syncthreads();
  // row softmax -> bf16 P
  if (lane < 16) {
    float mx = -3.0e38f;
    for (int k = 0; k < seqK; ++k) mx = fmaxf(mx, sS[lane][k]);
    float sum = 0.f;
    for (int k = 0; k < seqK; ++k) { float e = __expf(sS[lane][k] - mx); sS[lane][k] = e; sum += e; }
    float inv = 1.f / sum;
    for (int k = 0; k < seqK; ++k) sP[lane][k] = (bf16)(sS[lane][k] * inv);
  }
  __syncthreads();
  // O = P * V
  v8f o0 = vz, o1 = vz;
  const int kg = lane >> 4;
  for (int kc = 0; kc < seqK / 32; ++kc) {
    FragBF fa;
    fa.q[0] = *(const v4u*)&sP[nLoc][kc * 32 + kg * 16];
    fa.q[1] = *(const v4u*)&sP[nLoc][kc * 32 + kg * 16 + 8];
    v16bf bV0 = load_b_frag_rm(Vh, kc * 32, 0,  cHID);
    v16bf bV1 = load_b_frag_rm(Vh, kc * 32, 16, cHID);
    o0 = wmma_bf16(fa.v, bV0, o0);
    o1 = wmma_bf16(fa.v, bV1, o1);
  }
#pragma unroll
  for (int r = 0; r < 8; ++r) {
    int m = mBase + r;
    size_t rowOff = ((size_t)b * seqQ + q0 + m) * cHID + h * cDKH;
    O[rowOff + nLoc]      = (bf16)o0[r];
    O[rowOff + 16 + nLoc] = (bf16)o1[r];
  }
}

// ---------------- small scalar kernels ----------------
__global__ void cvt_bf16(const float* __restrict__ in, bf16* __restrict__ out, int n) {
  int i = blockIdx.x * blockDim.x + threadIdx.x;
  if (i < n) out[i] = (bf16)in[i];
}

// LayerNorm over 128 features, f32 in -> bf16 out. 1 wave per row, 4 rows per block.
__global__ void ln_kernel(const float* __restrict__ X, const float* __restrict__ g,
                          const float* __restrict__ bb, bf16* __restrict__ Y, int rows)
{
  int row  = blockIdx.x * 4 + (threadIdx.x >> 5);
  int lane = threadIdx.x & 31;
  if (row >= rows) return;
  const float* xr = X + (size_t)row * cHID;
  v4f x = *(const v4f*)(xr + lane * 4);
  float s  = x.x + x.y + x.z + x.w;
  float s2 = x.x * x.x + x.y * x.y + x.z * x.z + x.w * x.w;
#pragma unroll
  for (int m = 16; m >= 1; m >>= 1) {
    s  += __shfl_xor(s,  m, 32);
    s2 += __shfl_xor(s2, m, 32);
  }
  float mean = s * (1.f / cHID);
  float var  = s2 * (1.f / cHID) - mean * mean;
  float rstd = rsqrtf(var + 1e-5f);
  int c = lane * 4;
  bf16* yr = Y + (size_t)row * cHID + c;
  yr[0] = (bf16)((x.x - mean) * rstd * g[c + 0] + bb[c + 0]);
  yr[1] = (bf16)((x.y - mean) * rstd * g[c + 1] + bb[c + 1]);
  yr[2] = (bf16)((x.z - mean) * rstd * g[c + 2] + bb[c + 2]);
  yr[3] = (bf16)((x.w - mean) * rstd * g[c + 3] + bb[c + 3]);
}

// distance weights: dw[b,i,j] = (1/(dist+0.01)) normalized over i (column-normalized)
__global__ void dw_kernel(const float* __restrict__ center, float* __restrict__ dw) {
  int idx = blockIdx.x * blockDim.x + threadIdx.x;
  if (idx >= cB * cK) return;
  int b = idx >> 8, j = idx & 255;
  const float* cj = center + ((size_t)b * cK + j) * 3;
  float cx = cj[0], cy = cj[1], cz = cj[2];
  float sum = 0.f;
  for (int i = 0; i < cK; ++i) {
    const float* ci = center + ((size_t)b * cK + i) * 3;
    float dx = ci[0] - cx, dy = ci[1] - cy, dz = ci[2] - cz;
    sum += 1.f / (sqrtf(dx * dx + dy * dy + dz * dz) + 0.01f);
  }
  float inv = 1.f / sum;
  for (int i = 0; i < cK; ++i) {
    const float* ci = center + ((size_t)b * cK + i) * 3;
    float dx = ci[0] - cx, dy = ci[1] - cy, dz = ci[2] - cz;
    float w = 1.f / (sqrtf(dx * dx + dy * dy + dz * dz) + 0.01f);
    dw[((size_t)b * cK + i) * cK + j] = w * inv;
  }
}

// broadcast proposal features (B,K,HID) -> (B*CHUNK,K,HID)
__global__ void bcast_feats(const bf16* __restrict__ src, bf16* __restrict__ dst) {
  int i = blockIdx.x * blockDim.x + threadIdx.x;
  if (i >= M_FULL * cHID) return;
  int c   = i & 127;
  int row = i >> 7;
  int k   = row & 255;
  int bc  = row >> 8;
  int b   = bc / cCHUNK;
  dst[i] = src[((size_t)(b * cK + k)) * cHID + c];
}

// lang self-attention: h=4, dk=dv=16, width 64, seq 32. grid(BC, HEAD), block 32.
__global__ void lang_attn(const bf16* __restrict__ Q, const bf16* __restrict__ K,
                          const bf16* __restrict__ V, const int* __restrict__ mask,
                          bf16* __restrict__ O)
{
  int bc = blockIdx.x, h = blockIdx.y, q = threadIdx.x;
  const bf16* qp = Q + ((size_t)bc * cNW + q) * 64 + h * 16;
  float qv[16];
#pragma unroll
  for (int d = 0; d < 16; ++d) qv[d] = (float)qp[d];
  float s[32];
  float mx = -3.0e38f;
  for (int k = 0; k < cNW; ++k) {
    const bf16* kp = K + ((size_t)bc * cNW + k) * 64 + h * 16;
    float a = 0.f;
#pragma unroll
    for (int d = 0; d < 16; ++d) a += qv[d] * (float)kp[d];
    a *= 0.25f;  // 1/sqrt(16)
    if (!mask[bc * cNW + k]) a = NEGINF;
    s[k] = a;
    mx = fmaxf(mx, a);
  }
  float sum = 0.f;
  for (int k = 0; k < cNW; ++k) { s[k] = __expf(s[k] - mx); sum += s[k]; }
  float inv = 1.f / sum;
  float o[16];
#pragma unroll
  for (int d = 0; d < 16; ++d) o[d] = 0.f;
  for (int k = 0; k < cNW; ++k) {
    float p = s[k] * inv;
    const bf16* vp = V + ((size_t)bc * cNW + k) * 64 + h * 16;
#pragma unroll
    for (int d = 0; d < 16; ++d) o[d] += p * (float)vp[d];
  }
  bf16* op = O + ((size_t)bc * cNW + q) * 64 + h * 16;
#pragma unroll
  for (int d = 0; d < 16; ++d) op[d] = (bf16)o[d];
}

// final match projection to 1 channel
__global__ void match_out(const bf16* __restrict__ Y, const float* __restrict__ w3,
                          const float* __restrict__ b3, float* __restrict__ out)
{
  int row = blockIdx.x * blockDim.x + threadIdx.x;
  if (row >= M_FULL) return;
  const bf16* yr = Y + (size_t)row * cHID;
  float s = b3[0];
  for (int c = 0; c < cHID; ++c) s += (float)yr[c] * w3[c];
  out[row] = s;
}

// ---------------- host orchestration ----------------
extern "C" void kernel_launch(void* const* d_in, const int* in_sizes, int n_in,
                              void* d_out, int out_size, void* d_ws, size_t ws_size,
                              hipStream_t stream)
{
  (void)in_sizes; (void)n_in; (void)out_size; (void)ws_size;
  const float* in_pf = (const float*)d_in[0];   // proposal_feats (B,K,DET)
  const float* in_pc = (const float*)d_in[1];   // proposal_center (B,K,3)
  const float* in_lh = (const float*)d_in[3];   // lang_hiddens (BC,NW,LANG)
  const int*   in_lm = (const int*)d_in[4];     // lang_masks (BC,NW)

  // flattened param indices (setup_inputs dict order)
  enum {
    P_FC_W1 = 5, P_FC_B1, P_FC_G, P_FC_Bb, P_FC_M, P_FC_V, P_FC_A1, P_FC_W2, P_FC_B2,
    P_LF_W, P_LF_B, P_LF_LNG, P_LF_LNB,
    P_LS_WQ, P_LS_BQ, P_LS_WK, P_LS_BK, P_LS_WV, P_LS_BV, P_LS_WO, P_LS_BO, P_LS_LNG, P_LS_LNB,
    P_S0 = 28, P_S1 = 38, P_C0 = 48, P_C1 = 58, P_M = 68
  };
  auto F = [&](int i) { return (const float*)d_in[i]; };

  // bump allocator on workspace
  char* wsBase = (char*)d_ws;
  size_t wsOff = 0;
  auto alloc = [&](size_t bytes) -> void* {
    wsOff = (wsOff + 255) & ~(size_t)255;
    void* p = wsBase + wsOff;
    wsOff += bytes;
    return p;
  };
  auto cvtW = [&](int idx, size_t n) -> bf16* {
    bf16* dst = (bf16*)alloc(n * sizeof(bf16));
    cvt_bf16<<<((int)n + 255) / 256, 256, 0, stream>>>(F(idx), dst, (int)n);
    return dst;
  };

  // ---- convert all GEMM weights to bf16 ----
  bf16* Wfc1 = cvtW(P_FC_W1, (size_t)cHID * cDET);
  bf16* Wfc2 = cvtW(P_FC_W2, (size_t)cHID * cHID);
  bf16* Wlf  = cvtW(P_LF_W,  (size_t)cHID * cLANG);
  bf16* Wlsq = cvtW(P_LS_WQ, 64 * 128);
  bf16* Wlsk = cvtW(P_LS_WK, 64 * 128);
  bf16* Wlsv = cvtW(P_LS_WV, 64 * 128);
  bf16* Wlso = cvtW(P_LS_WO, 128 * 64);
  const int mhaBase[4] = {P_S0, P_S1, P_C0, P_C1};
  bf16* Wmha[4][4];
  for (int i = 0; i < 4; ++i)
    for (int j = 0; j < 4; ++j)
      Wmha[i][j] = cvtW(mhaBase[i] + 2 * j, 16384);
  bf16* Wm1 = cvtW(P_M + 0, 16384);
  bf16* Wm2 = cvtW(P_M + 7, 16384);

  // ---- activation buffers ----
  bf16*  aFeats = (bf16*)alloc((size_t)M_PROP * cDET * 2);
  bf16*  t1     = (bf16*)alloc((size_t)M_PROP * cHID * 2);
  bf16*  x0s    = (bf16*)alloc((size_t)M_PROP * cHID * 2);
  float* dwB    = (float*)alloc((size_t)cB * cK * cK * 4);
  bf16*  BQ  = (bf16*)alloc((size_t)M_FULL * cHID * 2);
  bf16*  BK  = (bf16*)alloc((size_t)M_FULL * cHID * 2);
  bf16*  BV  = (bf16*)alloc((size_t)M_FULL * cHID * 2);
  bf16*  BO  = (bf16*)alloc((size_t)M_FULL * cHID * 2);
  bf16*  BX0 = (bf16*)alloc((size_t)M_FULL * cHID * 2);
  bf16*  BX1 = (bf16*)alloc((size_t)M_FULL * cHID * 2);
  float* BF  = (float*)alloc((size_t)M_FULL * cHID * 4);
  bf16*  BL0 = (bf16*)alloc((size_t)M_LANG * cHID * 2);
  bf16*  BL1 = (bf16*)alloc((size_t)M_LANG * cHID * 2);

  auto gemm = [&](const bf16* A, const bf16* W, const float* bias, int M, int N, int Kd,
                  bf16* outb, float* outf, const bf16* resid,
                  const float* g, const float* bb, const float* mm, const float* vv,
                  const float* al, int alS, int mode) {
    dim3 grid(M / 64, N / 64);
    gemm_wmma<<<grid, 128, 0, stream>>>(A, W, bias, M, N, Kd, outb, outf, resid,
                                        g, bb, mm, vv, al, alS, mode);
  };

  // full MHA (h=4, dk=32) with residual + LayerNorm
  auto mha = [&](int pb, const bf16* const* Wset, const bf16* Xq, const bf16* Xkv,
                 int nBatch, int seqQ, int seqK,
                 const float* dwp, int dwDiv, const int* maskp, bf16* outX) {
    const int MQ = nBatch * seqQ, MK = nBatch * seqK;
    gemm(Xq,  Wset[0], F(pb + 1), MQ, cHID, cHID, BQ, nullptr, nullptr,
         nullptr, nullptr, nullptr, nullptr, nullptr, 0, EPI_BF16);
    gemm(Xkv, Wset[1], F(pb + 3), MK, cHID, cHID, BK, nullptr, nullptr,
         nullptr, nullptr, nullptr, nullptr, nullptr, 0, EPI_BF16);
    gemm(Xkv, Wset[2], F(pb + 5), MK, cHID, cHID, BV, nullptr, nullptr,
         nullptr, nullptr, nullptr, nullptr, nullptr, 0, EPI_BF16);
    attn_wmma<<<dim3(seqQ / 16, cHEAD, nBatch), 32, 0, stream>>>(
        BQ, BK, BV, BO, seqQ, seqK, dwp, dwDiv, maskp, 0.17677669529663687f);
    gemm(BO, Wset[3], F(pb + 7), MQ, cHID, cHID, nullptr, BF, Xq,
         nullptr, nullptr, nullptr, nullptr, nullptr, 0, EPI_F32_RES);
    ln_kernel<<<MQ / 4, 128, 0, stream>>>(BF, F(pb + 8), F(pb + 9), outX, MQ);
  };

  // ---- 1. features_concat: conv1 -> BN -> PReLU -> conv2 ----
  cvt_bf16<<<(M_PROP * cDET + 255) / 256, 256, 0, stream>>>(in_pf, aFeats, M_PROP * cDET);
  gemm(aFeats, Wfc1, F(P_FC_B1), M_PROP, cHID, cDET, t1, nullptr, nullptr,
       F(P_FC_G), F(P_FC_Bb), F(P_FC_M), F(P_FC_V), F(P_FC_A1), 1, EPI_BN_PRELU);
  gemm(t1, Wfc2, F(P_FC_B2), M_PROP, cHID, cHID, x0s, nullptr, nullptr,
       nullptr, nullptr, nullptr, nullptr, nullptr, 0, EPI_BF16);

  // ---- 2. distance bias ----
  dw_kernel<<<(cB * cK + 127) / 128, 128, 0, stream>>>(in_pc, dwB);

  // ---- 3. self0 over proposals (batch B, additive dw) ----
  mha(P_S0, Wmha[0], x0s, x0s, cB, cK, cK, dwB, 1, nullptr, BX0);

  // ---- 4. broadcast to per-description batch ----
  bcast_feats<<<((long long)M_FULL * cHID + 255) / 256, 256, 0, stream>>>(BX0, BX1);

  // ---- 5. language branch ----
  bf16* lhb = BV;  // reuse
  cvt_bf16<<<(M_LANG * cLANG + 255) / 256, 256, 0, stream>>>(in_lh, lhb, M_LANG * cLANG);
  gemm(lhb, Wlf, F(P_LF_B), M_LANG, cHID, cLANG, nullptr, BF, nullptr,
       nullptr, nullptr, nullptr, nullptr, nullptr, 0, EPI_F32_RELU);
  ln_kernel<<<M_LANG / 4, 128, 0, stream>>>(BF, F(P_LF_LNG), F(P_LF_LNB), BL0, M_LANG);
  gemm(BL0, Wlsq, F(P_LS_BQ), M_LANG, 64, cHID, BQ, nullptr, nullptr,
       nullptr, nullptr, nullptr, nullptr, nullptr, 0, EPI_BF16);
  gemm(BL0, Wlsk, F(P_LS_BK), M_LANG, 64, cHID, BK, nullptr, nullptr,
       nullptr, nullptr, nullptr, nullptr, nullptr, 0, EPI_BF16);
  gemm(BL0, Wlsv, F(P_LS_BV), M_LANG, 64, cHID, BV, nullptr, nullptr,
       nullptr, nullptr, nullptr, nullptr, nullptr, 0, EPI_BF16);
  lang_attn<<<dim3(cBC, cHEAD), cNW, 0, stream>>>(BQ, BK, BV, in_lm, BO);
  gemm(BO, Wlso, F(P_LS_BO), M_LANG, cHID, 64, nullptr, BF, BL0,
       nullptr, nullptr, nullptr, nullptr, nullptr, 0, EPI_F32_RES);
  ln_kernel<<<M_LANG / 4, 128, 0, stream>>>(BF, F(P_LS_LNG), F(P_LS_LNB), BL1, M_LANG);

  // ---- 6..8. cross0 -> self1 (+dw) -> cross1 ----
  mha(P_C0, Wmha[2], BX1, BL1, cBC, cK, cNW, nullptr, 1, in_lm, BX0);
  mha(P_S1, Wmha[1], BX0, BX0, cBC, cK, cK, dwB, cCHUNK, nullptr, BX1);
  mha(P_C1, Wmha[3], BX1, BL1, cBC, cK, cNW, nullptr, 1, in_lm, BX0);

  // ---- 9. match head ----
  gemm(BX0, Wm1, F(P_M + 1), M_FULL, cHID, cHID, BQ, nullptr, nullptr,
       F(P_M + 2), F(P_M + 3), F(P_M + 4), F(P_M + 5), F(P_M + 6), 0, EPI_BN_PRELU);
  gemm(BQ, Wm2, F(P_M + 8), M_FULL, cHID, cHID, BK, nullptr, nullptr,
       F(P_M + 9), F(P_M + 10), F(P_M + 11), F(P_M + 12), F(P_M + 13), 0, EPI_BN_PRELU);
  match_out<<<(M_FULL + 255) / 256, 256, 0, stream>>>(BK, F(P_M + 14), F(P_M + 15),
                                                      (float*)d_out);
}
